// Cross_Mamba_Attention_49263274885532
// MI455X (gfx1250) — matrix operand order repaired and checked by reading, not compiled, not run on hardware
//
#include <hip/hip_runtime.h>
#include <hip/hip_bf16.h>
#include <math.h>

typedef __attribute__((ext_vector_type(16))) _Float16 v16h;
typedef __attribute__((ext_vector_type(8)))  float    v8f;
typedef __attribute__((ext_vector_type(4)))  float    v4f;

#define EPSV   1e-5f
#define SCALEV 0.0625f   // 256^-0.5

#define Bsz  4
#define Lseq 4096
#define Dd   256
#define DIc  512
#define DSs  16
#define DCc  4
#define NTOK (Bsz*Lseq)   // 16384

// ---------------------------------------------------------------------------
// Batched GEMM: C = act(alpha * op(A) @ B + bias),  f16 WMMA frags, f32 accum.
//   op(A)(m,k) = TRANSA ? A[k*LDA_T+m] : A[m*lda+k]
//   B row-major (K x N) with compile-time LDB -> immediate-offset loads.
// One wave computes a 32x32 tile as 2x2 WMMA subtiles (4 v_wmma per K-chunk).
// 4 waves / 128-thread block. mode: 0=none, 1=+bias, 2=+bias,softplus.
// ---------------------------------------------------------------------------
template <int TRANSA, int KGUARD, int LDB, int LDA_T>
__global__ __launch_bounds__(128)
void wmma_gemm_kernel(const float* __restrict__ A, const float* __restrict__ B,
                      float* __restrict__ C, const float* __restrict__ bias,
                      int M, int N, int K, int lda, int ldc,
                      long long sA, long long sB, long long sC,
                      float alpha, int mode)
{
    const int z = blockIdx.z;
    A += (long long)z * sA;
    B += (long long)z * sB;
    C += (long long)z * sC;

    const int Nt    = N >> 5;
    const int tiles = (M >> 5) * Nt;
    const int w = blockIdx.x * 4 + (threadIdx.x >> 5);
    if (w >= tiles) return;                  // wave-uniform guard

    const int tm   = (w / Nt) * 32;
    const int tn   = (w % Nt) * 32;
    const int lane = threadIdx.x & 31;
    const int lrow = lane & 15;
    const int hi   = lane >> 4;              // 0 or 1
    const int kbA  = hi ? 8 : 0;             // A-frag K base (ISA 16-bit A layout)
    const int kbB  = hi ? 8 : 0;             // B-frag K base

    // Per-lane base pointers (computed once)
    const float* Ar0;
    const float* Ar1;
    if constexpr (TRANSA) {
        Ar0 = A + (tm + lrow);
        Ar1 = A + (tm + 16 + lrow);
    } else {
        Ar0 = A + (size_t)(tm + lrow) * lda;
        Ar1 = A + (size_t)(tm + 16 + lrow) * lda;
    }
    const float* Bc0 = B + (tn + lrow);
    const float* Bc1 = B + (tn + 16 + lrow);

    v8f acc00 = {}, acc01 = {}, acc10 = {}, acc11 = {};

    for (int kk = 0; kk < K; kk += 32) {
        v16h a0, a1, b0, b1;

        if constexpr (!TRANSA) {
            // Contiguous K: two 8-float runs per frag -> 4x global_load_b128
            if constexpr (!KGUARD) {
                const v4f* p0 = (const v4f*)(Ar0 + kk + kbA);
                const v4f* q0 = (const v4f*)(Ar0 + kk + kbA + 16);
                const v4f* p1 = (const v4f*)(Ar1 + kk + kbA);
                const v4f* q1 = (const v4f*)(Ar1 + kk + kbA + 16);
                v4f r0 = p0[0], r1 = p0[1], r2 = q0[0], r3 = q0[1];
                v4f s0 = p1[0], s1 = p1[1], s2 = q1[0], s3 = q1[1];
#pragma unroll
                for (int j = 0; j < 4; ++j) {
                    a0[j] = (_Float16)r0[j]; a0[4 + j]  = (_Float16)r1[j];
                    a0[8 + j] = (_Float16)r2[j]; a0[12 + j] = (_Float16)r3[j];
                    a1[j] = (_Float16)s0[j]; a1[4 + j]  = (_Float16)s1[j];
                    a1[8 + j] = (_Float16)s2[j]; a1[12 + j] = (_Float16)s3[j];
                }
            } else {
#pragma unroll
                for (int h = 0; h < 16; ++h) {
                    int k = kk + kbA + (h & 7) + ((h >> 3) << 4);
                    a0[h] = (_Float16)((k < K) ? Ar0[k] : 0.f);
                    a1[h] = (_Float16)((k < K) ? Ar1[k] : 0.f);
                }
            }
        } else {
            // Strided K: immediate offsets k*LDA_T off per-chunk base
            const float* pa0 = Ar0 + (size_t)kk * LDA_T;
            const float* pa1 = Ar1 + (size_t)kk * LDA_T;
#pragma unroll
            for (int h = 0; h < 16; ++h) {
                int ki = kbA + (h & 7) + ((h >> 3) << 4);
                if constexpr (KGUARD) {
                    a0[h] = (_Float16)((kk + ki < K) ? pa0[(size_t)ki * LDA_T] : 0.f);
                    a1[h] = (_Float16)((kk + ki < K) ? pa1[(size_t)ki * LDA_T] : 0.f);
                } else {
                    a0[h] = (_Float16)pa0[(size_t)ki * LDA_T];
                    a1[h] = (_Float16)pa1[(size_t)ki * LDA_T];
                }
            }
        }

        // B frags: base advanced per chunk, constant offsets kb*LDB
        {
            const float* pb0 = Bc0 + (size_t)(kk + kbB) * LDB;
            const float* pb1 = Bc1 + (size_t)(kk + kbB) * LDB;
#pragma unroll
            for (int h = 0; h < 16; ++h) {
                const int kb = (h & 7) + ((h >> 3) << 4);
                if constexpr (KGUARD) {
                    b0[h] = (_Float16)((kk + kbB + kb < K) ? pb0[(size_t)kb * LDB] : 0.f);
                    b1[h] = (_Float16)((kk + kbB + kb < K) ? pb1[(size_t)kb * LDB] : 0.f);
                } else {
                    b0[h] = (_Float16)pb0[(size_t)kb * LDB];
                    b1[h] = (_Float16)pb1[(size_t)kb * LDB];
                }
            }
        }

        if (kk + 32 < K) {   // gfx1250 global_prefetch_b8 for next chunk
            if constexpr (!TRANSA)
                __builtin_prefetch((const void*)(Ar0 + kk + 32 + kbA), 0, 3);
            else
                __builtin_prefetch((const void*)(Ar0 + (size_t)(kk + 32) * LDA_T), 0, 3);
            __builtin_prefetch((const void*)(Bc0 + (size_t)(kk + 32 + kbB) * LDB), 0, 3);
        }

        acc00 = __builtin_amdgcn_wmma_f32_16x16x32_f16(false, a0, false, b0,
                                                       (short)0, acc00, false, false);
        acc01 = __builtin_amdgcn_wmma_f32_16x16x32_f16(false, a0, false, b1,
                                                       (short)0, acc01, false, false);
        acc10 = __builtin_amdgcn_wmma_f32_16x16x32_f16(false, a1, false, b0,
                                                       (short)0, acc10, false, false);
        acc11 = __builtin_amdgcn_wmma_f32_16x16x32_f16(false, a1, false, b1,
                                                       (short)0, acc11, false, false);
    }

    // C/D layout: VGPR r -> M = r (lanes 0-15) / 8+r (lanes 16-31), N = lane&15
#pragma unroll
    for (int ms = 0; ms < 2; ++ms) {
#pragma unroll
        for (int ns = 0; ns < 2; ++ns) {
            const v8f acc = (ms == 0) ? (ns == 0 ? acc00 : acc01)
                                      : (ns == 0 ? acc10 : acc11);
            const int col = tn + ns * 16 + lrow;
            const float bv = (mode >= 1) ? bias[col] : 0.f;
#pragma unroll
            for (int r = 0; r < 8; ++r) {
                int row = tm + ms * 16 + r + (hi ? 8 : 0);
                float v = alpha * acc[r] + bv;
                if (mode == 2) v = (v > 20.f) ? v : log1pf(__expf(v));  // softplus
                C[(size_t)row * ldc + col] = v;
            }
        }
    }
}

// ---------------------------------------------------------------------------
// Row softmax over 256 columns, in place. One 256-thread block per row.
// ---------------------------------------------------------------------------
__global__ __launch_bounds__(256)
void softmax_rows(float* __restrict__ buf)
{
    __shared__ float red[256];
    const int row = blockIdx.x, t = threadIdx.x;
    float v = buf[(size_t)row * 256 + t];
    red[t] = v; __syncthreads();
    for (int o = 128; o > 0; o >>= 1) {
        if (t < o) red[t] = fmaxf(red[t], red[t + o]);
        __syncthreads();
    }
    const float mx = red[0];
    __syncthreads();
    const float e = __expf(v - mx);
    red[t] = e; __syncthreads();
    for (int o = 128; o > 0; o >>= 1) {
        if (t < o) red[t] += red[t + o];
        __syncthreads();
    }
    buf[(size_t)row * 256 + t] = e / red[0];
}

// ---------------------------------------------------------------------------
// hn = LayerNorm(2*ob) over last dim (256). One 256-thread block per row.
// ---------------------------------------------------------------------------
__global__ __launch_bounds__(256)
void ln2_rows(const float* __restrict__ ob, const float* __restrict__ w,
              const float* __restrict__ b, float* __restrict__ out)
{
    __shared__ float s1[256], s2[256];
    const int row = blockIdx.x, t = threadIdx.x;
    const float v = 2.f * ob[(size_t)row * 256 + t];
    s1[t] = v; s2[t] = v * v; __syncthreads();
    for (int o = 128; o > 0; o >>= 1) {
        if (t < o) { s1[t] += s1[t + o]; s2[t] += s2[t + o]; }
        __syncthreads();
    }
    const float mean = s1[0] * (1.f / 256.f);
    const float var  = s2[0] * (1.f / 256.f) - mean * mean;
    out[(size_t)row * 256 + t] = (v - mean) * rsqrtf(var + EPSV) * w[t] + b[t];
}

// ---------------------------------------------------------------------------
// Depthwise causal conv (kernel 4) + SiLU:  xs[b,l,d] = silu(conv(xi) + cb)
// ---------------------------------------------------------------------------
__global__ __launch_bounds__(256)
void conv_silu_kernel(const float* __restrict__ xz, const float* __restrict__ cw,
                      const float* __restrict__ cb, float* __restrict__ xs)
{
    const size_t idx = (size_t)blockIdx.x * 256 + threadIdx.x;
    const int    d   = (int)(idx & (DIc - 1));
    const size_t tok = idx >> 9;                 // 0..16383
    const int    l   = (int)(tok & (Lseq - 1));
    float acc = cb[d];
#pragma unroll
    for (int j = 0; j < DCc; ++j) {
        int ll = l - (DCc - 1) + j;
        if (ll >= 0)
            acc += cw[d * DCc + j] * xz[(tok - l + ll) * 1024 + d];
    }
    xs[tok * DIc + d] = acc / (1.f + __expf(-acc));   // silu
}

// ---------------------------------------------------------------------------
// Selective scan: per (b,d) channel, 16-state recurrence over L=4096.
// BC (32 floats per (b,l)) staged through LDS in chunks of 8 timesteps.
// ---------------------------------------------------------------------------
__global__ __launch_bounds__(256)
void mamba_scan(const float* __restrict__ dt, const float* __restrict__ xs,
                const float* __restrict__ BC, const float* __restrict__ xz,
                const float* __restrict__ A_log, const float* __restrict__ Dsk,
                float* __restrict__ y)
{
    const int d = blockIdx.x * 256 + threadIdx.x;   // 0..511
    const int b = blockIdx.y;

    float Ad[DSs];
#pragma unroll
    for (int s = 0; s < DSs; ++s) Ad[s] = -__expf(A_log[d * DSs + s]);
    const float Dv = Dsk[d];

    float h[DSs];
#pragma unroll
    for (int s = 0; s < DSs; ++s) h[s] = 0.f;

    __shared__ float sBC[8 * 32];
    const int tid = threadIdx.x;

    for (int l0 = 0; l0 < Lseq; l0 += 8) {
        sBC[tid] = BC[((size_t)(b * Lseq + l0 + (tid >> 5))) * 32 + (tid & 31)];
        __syncthreads();
#pragma unroll
        for (int j = 0; j < 8; ++j) {
            const size_t tok = (size_t)b * Lseq + l0 + j;
            const float dtv = dt[tok * DIc + d];
            const float xv  = xs[tok * DIc + d];
            float yv = 0.f;
#pragma unroll
            for (int s = 0; s < DSs; ++s) {
                const float dA = __expf(dtv * Ad[s]);
                h[s] = dA * h[s] + dtv * sBC[j * 32 + s] * xv;
                yv  += h[s] * sBC[j * 32 + 16 + s];
            }
            const float zv = xz[tok * 1024 + 512 + d];
            const float g  = zv / (1.f + __expf(-zv));     // silu gate
            y[tok * DIc + d] = (yv + xv * Dv) * g;
        }
        __syncthreads();
    }
}

// ---------------------------------------------------------------------------
template <int TRANSA, int KGUARD, int LDB, int LDA_T>
static void gemm(hipStream_t st, const float* A, const float* B, float* C,
                 const float* bias, int M, int N, int K, int lda, int ldc,
                 long long sA, long long sB, long long sC,
                 float alpha, int mode, int batch)
{
    const int tiles = (M / 32) * (N / 32);
    dim3 grid((tiles + 3) / 4, 1, batch);
    wmma_gemm_kernel<TRANSA, KGUARD, LDB, LDA_T>
        <<<grid, 128, 0, st>>>(A, B, C, bias, M, N, K, lda, ldc,
                               sA, sB, sC, alpha, mode);
}

extern "C" void kernel_launch(void* const* d_in, const int* in_sizes, int n_in,
                              void* d_out, int out_size, void* d_ws, size_t ws_size,
                              hipStream_t stream)
{
    (void)in_sizes; (void)n_in; (void)out_size; (void)ws_size;
    const float* x        = (const float*)d_in[0];
    const float* context  = (const float*)d_in[1];
    const float* Wq       = (const float*)d_in[2];
    const float* Wkv      = (const float*)d_in[3];
    const float* ln_w     = (const float*)d_in[4];
    const float* ln_b     = (const float*)d_in[5];
    const float* in_proj  = (const float*)d_in[6];
    const float* conv_w   = (const float*)d_in[7];
    const float* conv_b   = (const float*)d_in[8];
    const float* x_proj   = (const float*)d_in[9];
    const float* dt_proj  = (const float*)d_in[10];
    const float* dt_b     = (const float*)d_in[11];
    const float* A_log    = (const float*)d_in[12];
    const float* D_skip   = (const float*)d_in[13];
    const float* out_proj = (const float*)d_in[14];
    const float* Wout     = (const float*)d_in[15];
    const float* bout     = (const float*)d_in[16];
    float* out = (float*)d_out;
    float* ws  = (float*)d_ws;

    // Workspace regions (floats), lifetime-based reuse:
    //   R1 [0, 8.39M)   : kv (16384x512) -> ob (16384x256) -> xs (16384x512)
    //   R2 [8.39M,12.6M): {attn, W2} -> hn (16384x256) -> {BC, Wdt, Wo2}
    //   R3 [12.6M,29.4M): xz (16384x1024)
    //   R4 [29.4M,37.7M): dt (16384x512)
    //   R5 [37.7M,46.1M): y  (16384x512)
    float* kv   = ws;
    float* ob   = ws;
    float* xs   = ws;
    float* attn = ws + 8388608;
    float* W2   = attn + 262144;
    float* hn   = ws + 8388608;
    float* BCm  = ws + 8388608;
    float* Wdt  = BCm + 524288;
    float* Wo2  = Wdt + 262144;
    float* xzb  = ws + 12582912;
    float* dtb  = xzb + 16777216;
    float* yb   = dtb + 8388608;

    const long long sKV = (long long)Lseq * 512;
    const long long sDD = 65536;
    const long long sXB = (long long)Lseq * Dd;

    // 1) kv = context @ Wkv                               (16384 x 512, K=256)
    gemm<0,0,512,0>(stream, context, Wkv, kv, nullptr, NTOK, 512, 256, 256, 512,
                    0, 0, 0, 1.f, 0, 1);
    // 2) sim[b] = SCALE * k[b]^T @ v[b]                   (256 x 256, K=4096)
    gemm<1,0,512,512>(stream, kv, kv + 256, attn, nullptr, 256, 256, 4096, 512, 256,
                      sKV, sKV, sDD, SCALEV, 0, Bsz);
    // 3) attn = softmax_rows(sim)
    softmax_rows<<<dim3(Bsz * 256), 256, 0, stream>>>(attn);
    // 4) W2[b] = Wq @ attn[b]    (fold Wq:  ob = x @ (Wq@attn))
    gemm<0,0,256,0>(stream, Wq, attn, W2, nullptr, 256, 256, 256, 256, 256,
                    0, sDD, sDD, 1.f, 0, Bsz);
    // 5) ob[b] = x[b] @ W2[b]
    gemm<0,0,256,0>(stream, x, W2, ob, nullptr, Lseq, 256, 256, 256, 256,
                    sXB, sDD, sXB, 1.f, 0, Bsz);
    // 6) hn = LayerNorm(2*ob)
    ln2_rows<<<dim3(NTOK), 256, 0, stream>>>(ob, ln_w, ln_b, hn);
    // 7) xz = hn @ in_proj                                (16384 x 1024, K=256)
    gemm<0,0,1024,0>(stream, hn, in_proj, xzb, nullptr, NTOK, 1024, 256, 256, 1024,
                     0, 0, 0, 1.f, 0, 1);
    // 8) xs = silu(depthwise_causal_conv(xi) + conv_b)
    conv_silu_kernel<<<dim3(NTOK * DIc / 256), 256, 0, stream>>>(xzb, conv_w,
                                                                 conv_b, xs);
    // 9) Wdt = x_proj[:, :16] @ dt_proj   (fuse DR=16 bottleneck; K guarded)
    gemm<0,1,512,0>(stream, x_proj, dt_proj, Wdt, nullptr, 512, 512, 16, 48, 512,
                    0, 0, 0, 1.f, 0, 1);
    // 10) dt = softplus(xs @ Wdt + dt_proj_b)             (16384 x 512, K=512)
    gemm<0,0,512,0>(stream, xs, Wdt, dtb, dt_b, NTOK, 512, 512, 512, 512,
                    0, 0, 0, 1.f, 2, 1);
    // 11) BC = xs @ x_proj[:, 16:48]                      (16384 x 32,  K=512)
    gemm<0,0,48,0>(stream, xs, x_proj + 16, BCm, nullptr, NTOK, 32, 512, 512, 32,
                   0, 0, 0, 1.f, 0, 1);
    // 12) selective scan -> y (D-skip + silu(z) gate)
    mamba_scan<<<dim3(DIc / 256, Bsz), 256, 0, stream>>>(dtb, xs, BCm, xzb,
                                                         A_log, D_skip, yb);
    // 13) Wo2 = out_proj @ Wout   (fold output projections)
    gemm<0,0,256,0>(stream, out_proj, Wout, Wo2, nullptr, 512, 256, 256, 256, 256,
                    0, 0, 0, 1.f, 0, 1);
    // 14) out = y @ Wo2 + bout                            (16384 x 256, K=512)
    gemm<0,0,256,0>(stream, yb, Wo2, out, bout, NTOK, 256, 512, 512, 256,
                    0, 0, 0, 1.f, 1, 1);
}
